// sketchGenerator_42314017800367
// MI455X (gfx1250) — matrix operand order repaired and checked by reading, not compiled, not run on hardware
//
#include <hip/hip_runtime.h>

typedef __attribute__((ext_vector_type(16))) _Float16 v16h;
typedef __attribute__((ext_vector_type(8)))  _Float16 v8h;
typedef __attribute__((ext_vector_type(8)))  float    v8f;

#define B_  2048
#define T_  128
#define D_  5
#define H_  256
#define G4_ 1024

__device__ __forceinline__ float sigm(float x)      { return 1.0f / (1.0f + __expf(-x)); }
__device__ __forceinline__ float tanh_fast(float x) { return 2.0f / (1.0f + __expf(-2.0f * x)) - 1.0f; }
__device__ __forceinline__ float lrelu(float x)     { return x >= 0.0f ? x : 0.01f * x; }

// address of element (b, col) in WMMA-C-fragment-layout storage of a [B][H] matrix
__device__ __forceinline__ int cfrag_addr(int b, int col) {
    int Mt = b >> 4, Nt = col >> 4, m = b & 15, n = col & 15;
    int r = m & 7, hf = m >> 3;
    return ((Mt * (H_ / 16) + Nt) * 8 + r) * 32 + hf * 16 + n;
}

__global__ void f32_to_f16_kernel(const float* __restrict__ src, _Float16* __restrict__ dst, int n) {
    int i = blockIdx.x * 256 + threadIdx.x;
    if (i < n) dst[i] = (_Float16)src[i];
}

__global__ void bias_combine_kernel(const float* __restrict__ a, const float* __restrict__ b,
                                    float* __restrict__ o, int n) {
    int i = blockIdx.x * 256 + threadIdx.x;
    if (i < n) o[i] = a[i] + b[i];
}

// One LSTM timestep, fused: gates = h@Whh.T (WMMA f16) + x@Wih.T (VALU, D=5) + bias,
// then i/f/g/o nonlinearity, c update (fragment-layout, in place), h_out (f16 row-major).
// Grid: 256 blocks (16 btiles of 128 rows x 16 ntiles of 16 cols), 128 threads (4 waves).
// Each wave: 32 rows x 16 hidden cols, all 4 gates.
__global__ __launch_bounds__(128) void lstm_step_kernel(
    const _Float16* __restrict__ h_in,   // [B][H] f16
    _Float16* __restrict__ h_out,        // [B][H] f16
    float* __restrict__ c_frag,          // fragment layout, in-place update
    const _Float16* __restrict__ Whh,    // [4H][H] f16 (row g = W_hh[g,:])
    const float* __restrict__ Wih,       // [4H][D] f32
    const float* __restrict__ bias,      // [4H] combined b_ih+b_hh
    const float* __restrict__ x_base,    // rows of length D with stride xstride
    int xstride)
{
    __shared__ float sx[128 * D_];
    const int tid  = threadIdx.x;
    const int w    = tid >> 5;
    const int lane = tid & 31;
    const int half = lane >> 4;
    const int nlan = lane & 15;

    const int btile    = blockIdx.x >> 4;    // 0..15 -> 128 rows
    const int ntile    = blockIdx.x & 15;    // 0..15 -> 16 cols
    const int rowblock = btile * 128;
    const int colbase  = ntile * 16;

    // cooperative load of this block's 128 x rows (uniform trip count: 5 per thread)
    for (int i = tid; i < 128 * D_; i += 128) {
        int row = i / D_, d = i - row * D_;
        sx[i] = x_base[(size_t)(rowblock + row) * xstride + d];
    }
    __syncthreads();

    // per-lane input-weight columns and bias for the 4 gates
    const int jc = colbase + nlan;
    float wihv[4][D_];
    float bv[4];
#pragma unroll
    for (int G = 0; G < 4; ++G) {
        int g = G * H_ + jc;
        bv[G] = bias[g];
#pragma unroll
        for (int d = 0; d < D_; ++d) wihv[G][d] = Wih[g * D_ + d];
    }

    const int rowbase = rowblock + w * 32;
    const int mrow    = lane & 15;

    v8f zero = {};
    v8f acc[2][4];
#pragma unroll
    for (int s = 0; s < 2; ++s)
#pragma unroll
        for (int G = 0; G < 4; ++G) acc[s][G] = zero;

#pragma unroll
    for (int kk = 0; kk < H_; kk += 32) {
        // A fragments: contiguous k order, same as B. lanes0-15: K [kk,kk+16);
        // lanes16-31: K [kk+16,kk+32).
        v16h A[2];
#pragma unroll
        for (int s = 0; s < 2; ++s) {
            const _Float16* ap = h_in + (size_t)(rowbase + s * 16 + mrow) * H_ + kk + half * 16;
            v8h alo = *(const v8h*)ap;
            v8h ahi = *(const v8h*)(ap + 8);
#pragma unroll
            for (int i = 0; i < 8; ++i) { A[s][i] = alo[i]; A[s][i + 8] = ahi[i]; }
        }
        // B fragments: contiguous k order, same as A: lane n<16 -> col n, K [kk,kk+16);
        // lane>=16 -> col n-16, K [kk+16,kk+32). Column jc of Whh.T = row jc of Whh.
#pragma unroll
        for (int G = 0; G < 4; ++G) {
            const _Float16* bp = Whh + (size_t)(G * H_ + jc) * H_ + kk + half * 16;
            v16h Bf = *(const v16h*)bp;
            acc[0][G] = __builtin_amdgcn_wmma_f32_16x16x32_f16(
                false, A[0], false, Bf, (short)0, acc[0][G], false, false);
            acc[1][G] = __builtin_amdgcn_wmma_f32_16x16x32_f16(
                false, A[1], false, Bf, (short)0, acc[1][G], false, false);
        }
    }

    // pointwise LSTM update on C-layout accumulators
#pragma unroll
    for (int s = 0; s < 2; ++s) {
        const int Mtile = (rowbase >> 4) + s;
        const int cbase = (Mtile * (H_ / 16) + ntile) * 256;
#pragma unroll
        for (int r = 0; r < 8; ++r) {
            const int m = r + half * 8;
            const float* xr = &sx[(w * 32 + s * 16 + m) * D_];
            float gv[4];
#pragma unroll
            for (int G = 0; G < 4; ++G) {
                float a = acc[s][G][r] + bv[G];
#pragma unroll
                for (int d = 0; d < D_; ++d) a += xr[d] * wihv[G][d];
                gv[G] = a;
            }
            float iv = sigm(gv[0]);
            float fv = sigm(gv[1]);
            float gg = tanh_fast(gv[2]);
            float ov = sigm(gv[3]);
            int caddr = cbase + r * 32 + lane;
            float cn = fv * c_frag[caddr] + iv * gg;
            c_frag[caddr] = cn;
            float hn = ov * tanh_fast(cn);
            h_out[(size_t)(rowbase + s * 16 + m) * H_ + jc] = (_Float16)hn;
        }
    }
}

// hidden/cell = lrelu([h_f,0] @ W_enc.T + b_enc): only first H columns of W_enc matter.
// Writes generator's initial h (f16 row-major) and c (fragment layout).
__global__ __launch_bounds__(256) void enc_linear_kernel(
    const _Float16* __restrict__ h_f,      // [B][H]
    const float* __restrict__ c_frag_in,   // fragment layout
    const float* __restrict__ W_enc,       // [H][2H]
    const float* __restrict__ b_enc,       // [H]
    _Float16* __restrict__ h_gen,          // [B][H]
    float* __restrict__ c_frag_out)        // fragment layout
{
    __shared__ float sh[H_];
    __shared__ float sc[H_];
    int b = blockIdx.x, j = threadIdx.x;
    sh[j] = (float)h_f[(size_t)b * H_ + j];
    sc[j] = c_frag_in[cfrag_addr(b, j)];
    __syncthreads();
    float ah = 0.0f, ac = 0.0f;
    const float* wr = W_enc + (size_t)j * (2 * H_);
#pragma unroll 4
    for (int k = 0; k < H_; ++k) {
        float wv = wr[k];
        ah += sh[k] * wv;
        ac += sc[k] * wv;
    }
    float bb = b_enc[j];
    h_gen[(size_t)b * H_ + j] = (_Float16)lrelu(ah + bb);
    c_frag_out[cfrag_addr(b, j)] = lrelu(ac + bb);
}

// out = lrelu(h @ W_out.T + b_out) -> d_out[:,t,:] and next x. One wave per batch row.
__global__ __launch_bounds__(128) void gen_out_kernel(
    const _Float16* __restrict__ h,      // [B][H]
    const float* __restrict__ W_out,     // [D][H]
    const float* __restrict__ b_out,     // [D]
    float* __restrict__ out,             // [B][T][D]
    float* __restrict__ x_next,          // [B][D]
    int t)
{
    int w = threadIdx.x >> 5, lane = threadIdx.x & 31;
    int b = blockIdx.x * 4 + w;
    float acc[D_] = {0.0f, 0.0f, 0.0f, 0.0f, 0.0f};
#pragma unroll
    for (int i = 0; i < H_ / 32; ++i) {
        int k = lane + i * 32;
        float hv = (float)h[(size_t)b * H_ + k];
#pragma unroll
        for (int d = 0; d < D_; ++d) acc[d] += hv * W_out[d * H_ + k];
    }
#pragma unroll
    for (int off = 16; off > 0; off >>= 1)
#pragma unroll
        for (int d = 0; d < D_; ++d) acc[d] += __shfl_xor(acc[d], off, 32);
    if (lane == 0) {
#pragma unroll
        for (int d = 0; d < D_; ++d) {
            float v = lrelu(acc[d] + b_out[d]);
            out[((size_t)b * T_ + t) * D_ + d] = v;
            x_next[(size_t)b * D_ + d] = v;
        }
    }
}

extern "C" void kernel_launch(void* const* d_in, const int* in_sizes, int n_in,
                              void* d_out, int out_size, void* d_ws, size_t ws_size,
                              hipStream_t stream) {
    (void)in_sizes; (void)n_in; (void)out_size; (void)ws_size;

    const float* sketches = (const float*)d_in[0];
    const float* init_in  = (const float*)d_in[1];
    const float* W_ih     = (const float*)d_in[2];
    const float* W_hh     = (const float*)d_in[3];
    const float* b_ih     = (const float*)d_in[4];
    const float* b_hh     = (const float*)d_in[5];
    const float* W_ih_f   = (const float*)d_in[6];
    const float* W_hh_f   = (const float*)d_in[7];
    const float* b_ih_f   = (const float*)d_in[8];
    const float* b_hh_f   = (const float*)d_in[9];
    const float* W_enc    = (const float*)d_in[14];
    const float* b_enc    = (const float*)d_in[15];
    const float* W_out    = (const float*)d_in[16];
    const float* b_out    = (const float*)d_in[17];
    float* out = (float*)d_out;

    char* ws = (char*)d_ws;
    size_t off = 0;
    auto alloc = [&](size_t bytes) -> char* {
        char* p = ws + off;
        off += (bytes + 255) & ~(size_t)255;
        return p;
    };
    _Float16* Wh_enc = (_Float16*)alloc((size_t)G4_ * H_ * 2);
    _Float16* Wh_gen = (_Float16*)alloc((size_t)G4_ * H_ * 2);
    _Float16* h0     = (_Float16*)alloc((size_t)B_ * H_ * 2);
    _Float16* h1     = (_Float16*)alloc((size_t)B_ * H_ * 2);
    float*    c_enc  = (float*)alloc((size_t)B_ * H_ * 4);
    float*    c_gen  = (float*)alloc((size_t)B_ * H_ * 4);
    float*    bias_e = (float*)alloc((size_t)G4_ * 4);
    float*    bias_g = (float*)alloc((size_t)G4_ * 4);
    float*    x_gen  = (float*)alloc((size_t)B_ * D_ * 4);

    // setup: weight conversion, bias fusion, zero state
    f32_to_f16_kernel<<<(G4_ * H_ + 255) / 256, 256, 0, stream>>>(W_hh_f, Wh_enc, G4_ * H_);
    f32_to_f16_kernel<<<(G4_ * H_ + 255) / 256, 256, 0, stream>>>(W_hh,   Wh_gen, G4_ * H_);
    bias_combine_kernel<<<4, 256, 0, stream>>>(b_ih_f, b_hh_f, bias_e, G4_);
    bias_combine_kernel<<<4, 256, 0, stream>>>(b_ih,   b_hh,   bias_g, G4_);
    hipMemsetAsync(h0, 0, (size_t)B_ * H_ * 2, stream);
    hipMemsetAsync(c_enc, 0, (size_t)B_ * H_ * 4, stream);

    // encoder: forward LSTM over T steps (h ping-pongs h0<->h1; final h_f lands in h0)
    for (int t = 0; t < T_; ++t) {
        const _Float16* hin = (t & 1) ? h1 : h0;
        _Float16* hout      = (t & 1) ? h0 : h1;
        lstm_step_kernel<<<256, 128, 0, stream>>>(hin, hout, c_enc, Wh_enc, W_ih_f, bias_e,
                                                  sketches + (size_t)t * D_, T_ * D_);
    }

    // linear_encode: generator initial state (h into h1, c into c_gen)
    enc_linear_kernel<<<B_, 256, 0, stream>>>(h0, c_enc, W_enc, b_enc, h1, c_gen);

    // generator: autoregressive LSTM (h ping-pongs h1<->h0)
    for (int t = 0; t < T_; ++t) {
        const _Float16* hin = (t & 1) ? h0 : h1;
        _Float16* hout      = (t & 1) ? h1 : h0;
        const float* xp     = (t == 0) ? init_in : x_gen;
        lstm_step_kernel<<<256, 128, 0, stream>>>(hin, hout, c_gen, Wh_gen, W_ih, bias_g, xp, D_);
        gen_out_kernel<<<B_ / 4, 128, 0, stream>>>(hout, W_out, b_out, out, x_gen, t);
    }
}
